// Encoder_49014166782319
// MI455X (gfx1250) — matrix operand order repaired and checked by reading, not compiled, not run on hardware
//
#include <hip/hip_runtime.h>
#include <hip/hip_bf16.h>
#include <math.h>

#define NA    50000
#define MPAD  50016          // next multiple of 32 (pad rows for 32-row wave tiles)
#define NE    800000
#define NG    2
#define NEMB  128

typedef __attribute__((ext_vector_type(16))) __bf16 v16bf;
typedef __attribute__((ext_vector_type(8)))  float  v8f;

union Frag16 {
    v16bf v;
    uint4 q[2];
};

// ---------------------------------------------------------------------------
// Conversion / transpose helpers
// ---------------------------------------------------------------------------
__global__ void cvt_f32_bf16(const float* __restrict__ s, __bf16* __restrict__ d, int n) {
    int i = blockIdx.x * blockDim.x + threadIdx.x;
    if (i < n) d[i] = (__bf16)s[i];
}

// Wt[n*K + k] = W[k*N + n]  (W is [K_in][N_out] row-major)
__global__ void transpose_w_bf16(const float* __restrict__ W, __bf16* __restrict__ Wt,
                                 int K, int N) {
    int i = blockIdx.x * blockDim.x + threadIdx.x;
    if (i >= K * N) return;
    int n = i / K, k = i % K;
    Wt[i] = (__bf16)W[(size_t)k * N + n];
}

// x1f [NA][256] f32 -> z_bf columns [128, 384)
__global__ void cvt_x1_to_z(const float* __restrict__ x1f, __bf16* __restrict__ zbf) {
    int i = blockIdx.x * blockDim.x + threadIdx.x;
    if (i >= NA * 256) return;
    int r = i / 256, c = i % 256;
    zbf[(size_t)r * 384 + 128 + c] = (__bf16)x1f[i];
}

__global__ void attn_init(float* __restrict__ mmax, float* __restrict__ ssum,
                          float* __restrict__ x1f) {
    int i = blockIdx.x * blockDim.x + threadIdx.x;
    if (i < NG * NA) { mmax[i] = -INFINITY; ssum[i] = 0.0f; }
    if (i < NG * NA * NEMB) x1f[i] = 0.0f;
}

// ---------------------------------------------------------------------------
// bf16 WMMA GEMM: out[M,N] = act(A[M,K] @ W[K,N] + bias), W given as Wt[N][K].
// One wave computes a 32x64 tile: 2 A fragments x 4 B fragments -> 8 WMMAs
// per 32-deep k-step (12 b128 loads / 8 WMMAs). 8 waves per block.
// A-fragment layout (ISA 7.12.2, 16-bit A 16x32): lane L holds row m=L&15,
//   elems 0..7  = k in [kb + (L>>4)*8,      +8)
//   elems 8..15 = k in [kb + 16 + (L>>4)*8, +8)
// B-fragment (32x16): lane L holds col n=L&15, same k layout as A (elems 0..7 -> kb+(L>>4)*8+j, 8..15 -> kb+16+(L>>4)*8+j-8).
// C/D: elem i -> row m = i + 8*(L>>4), col n = L&15.
// A buffers are padded to >= 32-row multiples; stores guarded by m < M.
// ---------------------------------------------------------------------------
template<bool RELU, bool OUTBF16>
__global__ __launch_bounds__(256)
void gemm_bf16_wmma(const __bf16* __restrict__ A, int lda,
                    const __bf16* __restrict__ Bt,
                    const float* __restrict__ bias,
                    void* __restrict__ outp, int ldo,
                    int M, int N, int K) {
    int lane   = threadIdx.x & 31;
    int wave   = blockIdx.x * (blockDim.x >> 5) + (threadIdx.x >> 5);
    int mtiles = (M + 31) >> 5;
    int ngrp   = N >> 6;
    if (wave >= mtiles * ngrp) return;
    int r0 = (wave % mtiles) << 5;
    int c0 = (wave / mtiles) << 6;
    int half = lane >> 4;
    int lr   = lane & 15;

    const __bf16* arow0 = A + (size_t)(r0 + lr) * lda + half * 8;
    const __bf16* arow1 = arow0 + (size_t)16 * lda;
    const __bf16* b0    = Bt + (size_t)(c0 + lr) * K + half * 8;

    v8f acc[2][4];
    #pragma unroll
    for (int r = 0; r < 2; ++r)
        #pragma unroll
        for (int w = 0; w < 4; ++w)
            #pragma unroll
            for (int i = 0; i < 8; ++i) acc[r][w][i] = 0.0f;

    for (int kb = 0; kb < K; kb += 32) {
        Frag16 a0, a1;
        a0.q[0] = *(const uint4*)(arow0 + kb);
        a0.q[1] = *(const uint4*)(arow0 + kb + 16);
        a1.q[0] = *(const uint4*)(arow1 + kb);
        a1.q[1] = *(const uint4*)(arow1 + kb + 16);
        const __bf16* bptr = b0 + kb;
        #pragma unroll
        for (int w = 0; w < 4; ++w) {
            Frag16 b;
            b.q[0] = *(const uint4*)(bptr);
            b.q[1] = *(const uint4*)(bptr + 16);
            bptr += (size_t)16 * K;
            acc[0][w] = __builtin_amdgcn_wmma_f32_16x16x32_bf16(
                false, a0.v, false, b.v, (short)0, acc[0][w], false, false);
            acc[1][w] = __builtin_amdgcn_wmma_f32_16x16x32_bf16(
                false, a1.v, false, b.v, (short)0, acc[1][w], false, false);
        }
    }

    #pragma unroll
    for (int w = 0; w < 4; ++w) {
        int n = c0 + w * 16 + lr;
        float bv = bias[n];
        #pragma unroll
        for (int r = 0; r < 2; ++r) {
            #pragma unroll
            for (int i = 0; i < 8; ++i) {
                int m = r0 + r * 16 + i + 8 * half;
                if (m < M) {
                    float v = acc[r][w][i] + bv;
                    if (RELU) v = fmaxf(v, 0.0f);
                    if (OUTBF16) ((__bf16*)outp)[(size_t)m * ldo + n] = (__bf16)v;
                    else         ((float*)outp)[(size_t)m * ldo + n]  = v;
                }
            }
        }
    }
}

// ---------------------------------------------------------------------------
// Edge attention (per graph g = blockIdx.y), qkv layout: [NA][ g*384 + {q,k,v}*128 ]
// ---------------------------------------------------------------------------
__global__ __launch_bounds__(256)
void edge_score_max(const float* __restrict__ qkv, const int* __restrict__ ei,
                    float* __restrict__ score, float* __restrict__ mmax) {
    int g = blockIdx.y;
    int e = blockIdx.x * (blockDim.x >> 5) + (threadIdx.x >> 5);
    if (e >= NE) return;
    int lane = threadIdx.x & 31;
    int src = ei[(size_t)g * 2 * NE + e];
    int dst = ei[(size_t)g * 2 * NE + NE + e];
    const float4* qp = (const float4*)(qkv + (size_t)src * 768 + g * 384);
    const float4* kp = (const float4*)(qkv + (size_t)dst * 768 + g * 384 + 128);
    float4 qv = qp[lane], kv = kp[lane];
    float p = qv.x * kv.x + qv.y * kv.y + qv.z * kv.z + qv.w * kv.w;
    #pragma unroll
    for (int off = 16; off; off >>= 1) p += __shfl_xor(p, off, 32);
    if (lane == 0) {
        p *= 0.08838834764831845f;   // 1/sqrt(128)
        score[(size_t)g * NE + e] = p;
        atomicMax(&mmax[(size_t)g * NA + src], p);
    }
}

__global__ __launch_bounds__(256)
void edge_exp_sum(const int* __restrict__ ei, float* __restrict__ score,
                  const float* __restrict__ mmax, float* __restrict__ ssum) {
    int g = blockIdx.y;
    int e = blockIdx.x * blockDim.x + threadIdx.x;
    if (e >= NE) return;
    int src = ei[(size_t)g * 2 * NE + e];
    float w = __expf(score[(size_t)g * NE + e] - mmax[(size_t)g * NA + src]);
    score[(size_t)g * NE + e] = w;
    atomicAdd(&ssum[(size_t)g * NA + src], w);
}

__global__ __launch_bounds__(256)
void edge_scatter(const float* __restrict__ qkv, const int* __restrict__ ei,
                  const float* __restrict__ score, const float* __restrict__ ssum,
                  float* __restrict__ x1f) {
    int g = blockIdx.y;
    int e = blockIdx.x * (blockDim.x >> 5) + (threadIdx.x >> 5);
    if (e >= NE) return;
    int lane = threadIdx.x & 31;
    int src = ei[(size_t)g * 2 * NE + e];
    int dst = ei[(size_t)g * 2 * NE + NE + e];
    float attn = score[(size_t)g * NE + e] / ssum[(size_t)g * NA + src];
    const float4* vp = (const float4*)(qkv + (size_t)dst * 768 + g * 384 + 256);
    float4 vv = vp[lane];
    float* ob = x1f + (size_t)src * (NG * NEMB) + g * NEMB + lane * 4;
    atomicAdd(ob + 0, attn * vv.x);
    atomicAdd(ob + 1, attn * vv.y);
    atomicAdd(ob + 2, attn * vv.z);
    atomicAdd(ob + 3, attn * vv.w);
}

// ---------------------------------------------------------------------------
extern "C" void kernel_launch(void* const* d_in, const int* in_sizes, int n_in,
                              void* d_out, int out_size, void* d_ws, size_t ws_size,
                              hipStream_t stream) {
    const float* x     = (const float*)d_in[0];
    const int*   ei    = (const int*)d_in[1];
    const float* W_e1  = (const float*)d_in[2];
    const float* b_e1  = (const float*)d_in[3];
    const float* W_e2  = (const float*)d_in[4];
    const float* b_e2  = (const float*)d_in[5];
    const float* W_att = (const float*)d_in[6];
    const float* b_att = (const float*)d_in[7];
    const float* W_o1  = (const float*)d_in[8];
    const float* b_o1  = (const float*)d_in[9];
    const float* W_o2  = (const float*)d_in[10];
    const float* b_o2  = (const float*)d_in[11];
    float* out = (float*)d_out;

    char* ws = (char*)d_ws;
    size_t off = 0;
    auto alloc = [&](size_t bytes) -> void* {
        off = (off + 255) & ~(size_t)255;
        void* p = ws + off;
        off += bytes;
        return p;
    };
    // Activation buffers padded to MPAD rows (reads of pad rows are benign,
    // stores are guarded by m < M in the GEMM).
    __bf16* x_bf   = (__bf16*)alloc((size_t)MPAD * 128 * 2);
    __bf16* h1_bf  = (__bf16*)alloc((size_t)MPAD * 128 * 2);
    __bf16* z_bf   = (__bf16*)alloc((size_t)MPAD * 384 * 2);
    __bf16* zo_bf  = (__bf16*)alloc((size_t)MPAD * 128 * 2);
    float*  qkv    = (float*)alloc((size_t)NA * 768 * 4);
    float*  x1f    = (float*)alloc((size_t)NA * 256 * 4);
    float*  score  = (float*)alloc((size_t)NG * NE * 4);
    float*  mmax   = (float*)alloc((size_t)NG * NA * 4);
    float*  ssum   = (float*)alloc((size_t)NG * NA * 4);
    __bf16* Wt_e1  = (__bf16*)alloc(128 * 128 * 2);
    __bf16* Wt_e2  = (__bf16*)alloc(128 * 128 * 2);
    __bf16* Wt_att = (__bf16*)alloc(768 * 128 * 2);
    __bf16* Wt_o1  = (__bf16*)alloc(128 * 384 * 2);
    __bf16* Wt_o2  = (__bf16*)alloc(64 * 128 * 2);

    // Weight transposes + input conversion
    transpose_w_bf16<<<(128*128 + 255)/256, 256, 0, stream>>>(W_e1, Wt_e1, 128, 128);
    transpose_w_bf16<<<(128*128 + 255)/256, 256, 0, stream>>>(W_e2, Wt_e2, 128, 128);
    transpose_w_bf16<<<(128*768 + 255)/256, 256, 0, stream>>>(W_att, Wt_att, 128, 768);
    transpose_w_bf16<<<(384*128 + 255)/256, 256, 0, stream>>>(W_o1, Wt_o1, 384, 128);
    transpose_w_bf16<<<(128*64  + 255)/256, 256, 0, stream>>>(W_o2, Wt_o2, 128, 64);
    cvt_f32_bf16<<<(NA*128 + 255)/256, 256, 0, stream>>>(x, x_bf, NA * 128);

    auto gemm = [&](const __bf16* A, int lda, const __bf16* Bt, const float* bias,
                    void* o, int ldo, int M, int N, int K, bool relu, bool obf) {
        int waves = ((M + 31) >> 5) * (N >> 6);
        int blocks = (waves + 7) / 8;
        if (relu && obf)
            gemm_bf16_wmma<true, true><<<blocks, 256, 0, stream>>>(A, lda, Bt, bias, o, ldo, M, N, K);
        else if (relu && !obf)
            gemm_bf16_wmma<true, false><<<blocks, 256, 0, stream>>>(A, lda, Bt, bias, o, ldo, M, N, K);
        else if (!relu && obf)
            gemm_bf16_wmma<false, true><<<blocks, 256, 0, stream>>>(A, lda, Bt, bias, o, ldo, M, N, K);
        else
            gemm_bf16_wmma<false, false><<<blocks, 256, 0, stream>>>(A, lda, Bt, bias, o, ldo, M, N, K);
    };

    // h1 = relu(x @ W_e1 + b_e1)
    gemm(x_bf, 128, Wt_e1, b_e1, h1_bf, 128, NA, 128, 128, true, true);
    // h2 = relu(h1 @ W_e2 + b_e2) -> z columns [0,128)
    gemm(h1_bf, 128, Wt_e2, b_e2, z_bf, 384, NA, 128, 128, true, true);
    // qkv = h2 @ W_att + b_att (f32)
    gemm(z_bf, 384, Wt_att, b_att, qkv, 768, NA, 768, 128, false, false);

    // Edge attention
    attn_init<<<(NG*NA*NEMB + 255)/256, 256, 0, stream>>>(mmax, ssum, x1f);
    dim3 wgrid((NE + 7) / 8, NG);
    dim3 tgrid((NE + 255) / 256, NG);
    edge_score_max<<<wgrid, 256, 0, stream>>>(qkv, ei, score, mmax);
    edge_exp_sum<<<tgrid, 256, 0, stream>>>(ei, score, mmax, ssum);
    edge_scatter<<<wgrid, 256, 0, stream>>>(qkv, ei, score, ssum, x1f);
    cvt_x1_to_z<<<(NA*256 + 255)/256, 256, 0, stream>>>(x1f, z_bf);

    // z -> relu(z @ W_o1 + b_o1) -> out = zo @ W_o2 + b_o2
    gemm(z_bf, 384, Wt_o1, b_o1, zo_bf, 128, NA, 128, 384, true, true);
    gemm(zo_bf, 128, Wt_o2, b_o2, out, 64, NA, 64, 128, false, false);
}
